// SoftmaxLossRScore_42820823941334
// MI455X (gfx1250) — matrix operand order repaired and checked by reading, not compiled, not run on hardware
//
#include <hip/hip_runtime.h>

#define B_ROWS 2048
#define N_COLS 32768
#define D_K    128
#define NCHUNK 8
#define CHUNK_N (N_COLS / NCHUNK)       // 4096 columns per block
#define WAVES_PER_BLOCK 8
#define TILES_PER_CHUNK (CHUNK_N / 16)  // 256 tiles; 2 per wave per iteration

#define LOG2E_F 1.44269504088896340736f
#define LN2_F   0.69314718055994530942f
#define NEG_BIG (-3.0e38f)

typedef __bf16 v16bf  __attribute__((ext_vector_type(16)));
typedef __bf16 bf16x8 __attribute__((ext_vector_type(8)));
typedef float  v8f    __attribute__((ext_vector_type(8)));

__device__ __forceinline__ unsigned short f32_to_bf16_rne(float x) {
    unsigned u = __float_as_uint(x);
    unsigned r = u + 0x7FFFu + ((u >> 16) & 1u);
    return (unsigned short)(r >> 16);
}

// Merge two (max, sum) logsumexp states with a single exp2.
__device__ __forceinline__ void lse_merge(float& m1, float& s1, float m2, float s2) {
    float d = m2 - m1;
    float e = __builtin_amdgcn_exp2f(-__builtin_fabsf(d));   // |.| folds into exp input mods
    bool  up = d > 0.0f;
    s1 = up ? fmaf(s1, e, s2) : fmaf(s2, e, s1);
    m1 = up ? m2 : m1;
}

// Online LSE update with one score value, single exp2.
__device__ __forceinline__ void lse_update(float& m, float& s, float v) {
    float d = v - m;
    float e = __builtin_amdgcn_exp2f(-__builtin_fabsf(d));
    bool  up = d > 0.0f;
    float s_stay = s + e;                 // max unchanged: add 2^(v-m)
    float s_move = fmaf(s, e, 1.0f);      // new max = v: rescale, add 2^0
    s = up ? s_move : s_stay;
    m = up ? v : m;
}

// Split fp32 -> bf16 hi + bf16 residual (bf16x3 compensation), with optional
// pre-scale (anchor is pre-scaled by log2(e) so WMMA outputs base-2 scores).
__global__ void split_bf16_kernel(const float* __restrict__ in,
                                  unsigned short* __restrict__ hi,
                                  unsigned short* __restrict__ lo,
                                  float scale, int n) {
    int i = blockIdx.x * blockDim.x + threadIdx.x;
    if (i >= n) return;
    float x = in[i] * scale;
    unsigned short h = f32_to_bf16_rne(x);
    float hf = __uint_as_float(((unsigned)h) << 16);
    hi[i] = h;
    lo[i] = f32_to_bf16_rne(x - hf);
}

// Fused (anchor @ negative^T) + streaming row-wise logsumexp (base 2).
// Block: 16 rows x 4096 cols.  Wave: two 16x16 WMMA tiles per iteration
// (independent accumulator chains to keep the XDL pipe full and fill the
// WMMA->VALU hazard slots with useful work).
__global__ __launch_bounds__(256)
void gemm_lse_kernel(const __bf16* __restrict__ aHi, const __bf16* __restrict__ aLo,
                     const __bf16* __restrict__ bHi, const __bf16* __restrict__ bLo,
                     float* __restrict__ partials) {
    const int rowTile = blockIdx.x;   // 0..127
    const int chunk   = blockIdx.y;   // 0..NCHUNK-1
    const int tid  = threadIdx.x;
    const int wave = tid >> 5;
    const int lane = tid & 31;
    const int half = lane >> 4;
    const int l15  = lane & 15;

    union Frag { v16bf v; bf16x8 h[2]; };

    // ---- A fragments (16x32 bf16 per K-block), contiguous per-lane chunks ----
    const __bf16* aRowHi = aHi + (size_t)(rowTile * 16 + l15) * D_K;
    const __bf16* aRowLo = aLo + (size_t)(rowTile * 16 + l15) * D_K;
    Frag fa_hi[4], fa_lo[4];
#pragma unroll
    for (int kb = 0; kb < 4; ++kb) {
        const int k0 = kb * 32 + half * 16;
        fa_hi[kb].h[0] = *(const bf16x8*)(aRowHi + k0);
        fa_hi[kb].h[1] = *(const bf16x8*)(aRowHi + k0 + 8);
        fa_lo[kb].h[0] = *(const bf16x8*)(aRowLo + k0);
        fa_lo[kb].h[1] = *(const bf16x8*)(aRowLo + k0 + 8);
    }

    // Per-lane online LSE state: 8 rows (this lane's half), one column per lane.
    float m[8], s[8];
#pragma unroll
    for (int r = 0; r < 8; ++r) { m[r] = NEG_BIG; s[r] = 0.0f; }

    for (int t = 2 * wave; t < TILES_PER_CHUNK; t += 2 * WAVES_PER_BLOCK) {
        const int col0 = chunk * CHUNK_N + t * 16 + l15;  // negative rows this lane feeds
        const __bf16* b0Hi = bHi + (size_t)col0 * D_K;
        const __bf16* b0Lo = bLo + (size_t)col0 * D_K;
        const __bf16* b1Hi = b0Hi + (size_t)16 * D_K;     // tile t+1
        const __bf16* b1Lo = b0Lo + (size_t)16 * D_K;
        v8f c0 = {}, c1 = {};
#pragma unroll
        for (int kb = 0; kb < 4; ++kb) {
            const int k0 = kb * 32 + half * 16;           // B: 16 contiguous K per lane
            v16bf bh0 = *(const v16bf*)(b0Hi + k0);
            v16bf bl0 = *(const v16bf*)(b0Lo + k0);
            v16bf bh1 = *(const v16bf*)(b1Hi + k0);
            v16bf bl1 = *(const v16bf*)(b1Lo + k0);
            c0 = __builtin_amdgcn_wmma_f32_16x16x32_bf16(false, fa_hi[kb].v, false, bh0,
                                                         (short)0, c0, false, false);
            c1 = __builtin_amdgcn_wmma_f32_16x16x32_bf16(false, fa_hi[kb].v, false, bh1,
                                                         (short)0, c1, false, false);
            c0 = __builtin_amdgcn_wmma_f32_16x16x32_bf16(false, fa_hi[kb].v, false, bl0,
                                                         (short)0, c0, false, false);
            c1 = __builtin_amdgcn_wmma_f32_16x16x32_bf16(false, fa_hi[kb].v, false, bl1,
                                                         (short)0, c1, false, false);
            c0 = __builtin_amdgcn_wmma_f32_16x16x32_bf16(false, fa_lo[kb].v, false, bh0,
                                                         (short)0, c0, false, false);
            c1 = __builtin_amdgcn_wmma_f32_16x16x32_bf16(false, fa_lo[kb].v, false, bh1,
                                                         (short)0, c1, false, false);
        }
        // Online updates, one v_exp_f32 per value (co-executes with the XDL pipe).
#pragma unroll
        for (int r = 0; r < 8; ++r) lse_update(m[r], s[r], c0[r]);
#pragma unroll
        for (int r = 0; r < 8; ++r) lse_update(m[r], s[r], c1[r]);
    }

    // Butterfly-combine the 16 columns held by the 16 lanes of this half.
#pragma unroll
    for (int off = 8; off >= 1; off >>= 1) {
#pragma unroll
        for (int r = 0; r < 8; ++r) {
            float om = __shfl_xor(m[r], off, 32);
            float os = __shfl_xor(s[r], off, 32);
            lse_merge(m[r], s[r], om, os);
        }
    }

    __shared__ float lds_m[WAVES_PER_BLOCK][16];
    __shared__ float lds_s[WAVES_PER_BLOCK][16];
    if (l15 == 0) {
#pragma unroll
        for (int r = 0; r < 8; ++r) {
            lds_m[wave][half * 8 + r] = m[r];
            lds_s[wave][half * 8 + r] = s[r];
        }
    }
    __syncthreads();

    if (tid < 16) {   // combine the 8 waves, emit per-(row, chunk) partial
        float mm = lds_m[0][tid], ss = lds_s[0][tid];
#pragma unroll
        for (int w = 1; w < WAVES_PER_BLOCK; ++w)
            lse_merge(mm, ss, lds_m[w][tid], lds_s[w][tid]);
        const int row = rowTile * 16 + tid;
        partials[(row * NCHUNK + chunk) * 2 + 0] = mm;
        partials[(row * NCHUNK + chunk) * 2 + 1] = ss;
    }
}

// Per-row: merge chunk partials -> logsumexp; pos_sim dot; relu loss.
__global__ void row_loss_kernel(const float* __restrict__ partials,
                                const float* __restrict__ anchor,
                                const float* __restrict__ positive,
                                const float* __restrict__ r_score,
                                float* __restrict__ rowLoss) {
    int b = blockIdx.x * blockDim.x + threadIdx.x;
    if (b >= B_ROWS) return;
    float mm = NEG_BIG, ss = 0.0f;
#pragma unroll
    for (int c = 0; c < NCHUNK; ++c)
        lse_merge(mm, ss,
                  partials[(b * NCHUNK + c) * 2 + 0],
                  partials[(b * NCHUNK + c) * 2 + 1]);
    float lse = LN2_F * (mm + __builtin_amdgcn_logf(ss));   // natural-log LSE

    const float4* a4 = (const float4*)(anchor   + (size_t)b * D_K);
    const float4* p4 = (const float4*)(positive + (size_t)b * D_K);
    float pos = 0.0f;
#pragma unroll
    for (int i = 0; i < D_K / 4; ++i) {
        float4 a = a4[i], p = p4[i];
        pos += a.x * p.x + a.y * p.y + a.z * p.z + a.w * p.w;
    }
    float loss = lse - pos + r_score[b] + 1.0f;  // ALPHA = 1.0
    rowLoss[b] = fmaxf(loss, 0.0f);
}

// Deterministic fixed-order mean of the 2048 per-row losses.
__global__ void mean_kernel(const float* __restrict__ rowLoss, float* __restrict__ out) {
    __shared__ float sm[256];
    int tid = threadIdx.x;
    float acc = 0.0f;
    for (int i = tid; i < B_ROWS; i += 256) acc += rowLoss[i];
    sm[tid] = acc;
    __syncthreads();
    for (int s = 128; s > 0; s >>= 1) {
        if (tid < s) sm[tid] += sm[tid + s];
        __syncthreads();
    }
    if (tid == 0) out[0] = sm[0] * (1.0f / (float)B_ROWS);
}

extern "C" void kernel_launch(void* const* d_in, const int* in_sizes, int n_in,
                              void* d_out, int out_size, void* d_ws, size_t ws_size,
                              hipStream_t stream) {
    const float* anchor   = (const float*)d_in[0];
    const float* positive = (const float*)d_in[1];
    const float* negative = (const float*)d_in[2];
    const float* r_score  = (const float*)d_in[3];

    // Workspace layout (~17.2 MB): bf16 hi/lo copies + chunk partials + row losses.
    unsigned short* aHi = (unsigned short*)d_ws;
    unsigned short* aLo = aHi + (size_t)B_ROWS * D_K;
    unsigned short* nHi = aLo + (size_t)B_ROWS * D_K;
    unsigned short* nLo = nHi + (size_t)N_COLS * D_K;
    float* partials = (float*)(nLo + (size_t)N_COLS * D_K);
    float* rowLoss  = partials + B_ROWS * NCHUNK * 2;

    const int nA = B_ROWS * D_K;
    const int nN = N_COLS * D_K;
    // Anchor pre-scaled by log2(e): WMMA emits base-2 scores directly.
    split_bf16_kernel<<<(nA + 255) / 256, 256, 0, stream>>>(anchor, aHi, aLo, LOG2E_F, nA);
    split_bf16_kernel<<<(nN + 255) / 256, 256, 0, stream>>>(negative, nHi, nLo, 1.0f, nN);

    gemm_lse_kernel<<<dim3(B_ROWS / 16, NCHUNK), 256, 0, stream>>>(
        (const __bf16*)aHi, (const __bf16*)aLo,
        (const __bf16*)nHi, (const __bf16*)nLo, partials);

    row_loss_kernel<<<(B_ROWS + 255) / 256, 256, 0, stream>>>(
        partials, anchor, positive, r_score, rowLoss);
    mean_kernel<<<1, 256, 0, stream>>>(rowLoss, (float*)d_out);
}
